// MoE_28853590294770
// MI455X (gfx1250) — matrix operand order repaired and checked by reading, not compiled, not run on hardware
//
#include <hip/hip_runtime.h>
#include <math.h>

// MoE top-2 of 8 experts, D=1024, 8192 tokens.
// f16 WMMA (v_wmma_f32_16x16x32_f16) expert GEMMs, fp32 gating,
// async global->LDS double-buffered staging on CDNA5.

#define D_DIM   1024
#define E_NUM   8
#define TOKENS  8192
#define BM      64
#define BN      128
#define KC      64
#define LDT     72          // halves per LDS tile row (64 + 8 pad, 144B: 16B-aligned)
#define NSTAGE  (D_DIM / KC)

typedef _Float16 v4h  __attribute__((ext_vector_type(4)));
typedef _Float16 v8h  __attribute__((ext_vector_type(8)));
typedef _Float16 v16h __attribute__((ext_vector_type(16)));
typedef float    v8f  __attribute__((ext_vector_type(8)));
typedef int      v4i  __attribute__((ext_vector_type(4)));

#if defined(__AMDGCN__) && __has_builtin(__builtin_amdgcn_global_load_async_to_lds_b128)
#define HAS_ASYNC_LDS 1
#else
#define HAS_ASYNC_LDS 0
#endif

#if defined(__AMDGCN__) && __has_builtin(__builtin_amdgcn_s_wait_asynccnt)
#define WAIT_ASYNC() __builtin_amdgcn_s_wait_asynccnt(0)
#elif defined(__AMDGCN__)
#define WAIT_ASYNC() asm volatile("s_wait_asynccnt 0" ::: "memory")
#else
#define WAIT_ASYNC() do {} while (0)
#endif

#if HAS_ASYNC_LDS
__device__ __forceinline__ void async_cp16(const _Float16* g, _Float16* l) {
  // flat LDS address bits [31:0] are the LDS offset (ISA 10.2 aperture mapping)
  __builtin_amdgcn_global_load_async_to_lds_b128(
      (__attribute__((address_space(1))) v4i*)(unsigned long long)(uintptr_t)g,
      (__attribute__((address_space(3))) v4i*)(unsigned int)(uintptr_t)l,
      0, 0);
}
#endif

// ---------------- zero output + expert counters ----------------
__global__ __launch_bounds__(256) void moe_zero_kernel(float* __restrict__ out,
                                                       int n, int* __restrict__ counts) {
  int tid = blockIdx.x * blockDim.x + threadIdx.x;
  if (tid < E_NUM) counts[tid] = 0;
  for (int i = tid; i < n; i += gridDim.x * blockDim.x) out[i] = 0.0f;
}

// ---------------- x fp32 -> f16 ----------------
__global__ __launch_bounds__(256) void moe_conv_x(const float4* __restrict__ x,
                                                  v4h* __restrict__ xh, int n4) {
  int tid = blockIdx.x * blockDim.x + threadIdx.x;
  for (int i = tid; i < n4; i += gridDim.x * blockDim.x) {
    float4 v = x[i];
    v4h o;
    o[0] = (_Float16)v.x; o[1] = (_Float16)v.y;
    o[2] = (_Float16)v.z; o[3] = (_Float16)v.w;
    xh[i] = o;
  }
}

// ---------------- We fp32 [e][k][n] -> f16 transposed [e][n][k] ----------------
__global__ __launch_bounds__(256) void moe_conv_we(const float* __restrict__ We,
                                                   _Float16* __restrict__ Wt) {
  __shared__ float tile[32][33];
  int e  = blockIdx.z;
  int kb = blockIdx.y * 32;
  int nb = blockIdx.x * 32;
  int tx = threadIdx.x;        // 0..31
  int ty = threadIdx.y;        // 0..7
  const float* src = We + (size_t)e * D_DIM * D_DIM;
  _Float16*    dst = Wt + (size_t)e * D_DIM * D_DIM;
#pragma unroll
  for (int j = 0; j < 4; ++j)
    tile[ty + 8 * j][tx] = src[(size_t)(kb + ty + 8 * j) * D_DIM + nb + tx];
  __syncthreads();
#pragma unroll
  for (int j = 0; j < 4; ++j)
    dst[(size_t)(nb + ty + 8 * j) * D_DIM + kb + tx] = (_Float16)tile[tx][ty + 8 * j];
}

// ---------------- gating: scores, top-2, softmax, per-expert lists ----------------
__global__ __launch_bounds__(128) void moe_gate_kernel(const float* __restrict__ x,
                                                       const float* __restrict__ Wg,
                                                       const float* __restrict__ bg,
                                                       int* __restrict__ counts,
                                                       int* __restrict__ tok_ids,
                                                       float* __restrict__ tok_w) {
  int wib  = threadIdx.x >> 5;
  int lane = threadIdx.x & 31;
  int t    = blockIdx.x * 4 + wib;
  if (t >= TOKENS) return;

  float acc[E_NUM];
#pragma unroll
  for (int e = 0; e < E_NUM; ++e) acc[e] = 0.0f;

  const float* xr = x + (size_t)t * D_DIM;
  for (int d = lane; d < D_DIM; d += 32) {
    float xv = xr[d];
    const float4* wr = (const float4*)(Wg + (size_t)d * E_NUM);
    float4 w0 = wr[0], w1 = wr[1];
    acc[0] += xv * w0.x; acc[1] += xv * w0.y; acc[2] += xv * w0.z; acc[3] += xv * w0.w;
    acc[4] += xv * w1.x; acc[5] += xv * w1.y; acc[6] += xv * w1.z; acc[7] += xv * w1.w;
  }
#pragma unroll
  for (int off = 16; off > 0; off >>= 1) {
#pragma unroll
    for (int e = 0; e < E_NUM; ++e) acc[e] += __shfl_xor(acc[e], off, 32);
  }
  if (lane == 0) {
#pragma unroll
    for (int e = 0; e < E_NUM; ++e) acc[e] += bg[e];
    int   i0 = 0;       float v0 = acc[0];
#pragma unroll
    for (int e = 1; e < E_NUM; ++e) if (acc[e] > v0) { v0 = acc[e]; i0 = e; }
    int   i1 = -1;      float v1 = -3.4e38f;
#pragma unroll
    for (int e = 0; e < E_NUM; ++e) if (e != i0 && acc[e] > v1) { v1 = acc[e]; i1 = e; }
    float e1 = __expf(v1 - v0);
    float s  = 1.0f + e1;
    float w0 = 1.0f / s;
    float w1 = e1 / s;
    int slot0 = atomicAdd(&counts[i0], 1);
    tok_ids[i0 * TOKENS + slot0] = t;
    tok_w  [i0 * TOKENS + slot0] = w0;
    int slot1 = atomicAdd(&counts[i1], 1);
    tok_ids[i1 * TOKENS + slot1] = t;
    tok_w  [i1 * TOKENS + slot1] = w1;
  }
}

// ---------------- grouped expert GEMM with f16 WMMA ----------------
__global__ __launch_bounds__(256) void moe_gemm_kernel(const _Float16* __restrict__ xh,
                                                       const _Float16* __restrict__ wt,
                                                       const float* __restrict__ be,
                                                       const int* __restrict__ counts,
                                                       const int* __restrict__ tok_ids,
                                                       const float* __restrict__ tok_w,
                                                       float* __restrict__ out) {
  int e   = blockIdx.z;
  int cnt = counts[e];
  int mb  = blockIdx.y * BM;
  if (mb >= cnt) return;
  int nb  = blockIdx.x * BN;

  __shared__ __align__(16) _Float16 As[2][BM * LDT];
  __shared__ __align__(16) _Float16 Bs[2][BN * LDT];
  __shared__ int   s_tok[BM];
  __shared__ float s_w[BM];

  int tid = threadIdx.x;
  if (tid < BM) {
    int  gi = mb + tid;
    bool vl = gi < cnt;
    s_tok[tid] = vl ? tok_ids[e * TOKENS + gi] : -1;
    s_w  [tid] = vl ? tok_w  [e * TOKENS + gi] : 0.0f;
  }
  __syncthreads();

  // staging roles: 16B segments, 8 per tile row (KC=64 halves)
  int acs = (tid & 7) * 8;  // half offset within row
  int ar0 = tid >> 3;       // 0..31
  int ar1 = ar0 + 32;
  int t0 = s_tok[ar0]; if (t0 < 0) t0 = 0;   // invalid rows: garbage confined to unstored C rows
  int t1 = s_tok[ar1]; if (t1 < 0) t1 = 0;
  const _Float16* ag0 = xh + (size_t)t0 * D_DIM + acs;
  const _Float16* ag1 = xh + (size_t)t1 * D_DIM + acs;
  const _Float16* wb  = wt + (size_t)e * D_DIM * D_DIM;
  const _Float16* bg0 = wb + (size_t)(nb + ar0     ) * D_DIM + acs;
  const _Float16* bg1 = wb + (size_t)(nb + ar0 + 32) * D_DIM + acs;
  const _Float16* bg2 = wb + (size_t)(nb + ar0 + 64) * D_DIM + acs;
  const _Float16* bg3 = wb + (size_t)(nb + ar0 + 96) * D_DIM + acs;

  int wid  = tid >> 5;
  int lane = tid & 31;
  int wm   = (wid & 1) * 32;   // 0 or 32
  int wn   = (wid >> 1) * 32;  // 0,32,64,96
  int h    = lane >> 4;
  int nl   = lane & 15;

  v8f c00 = {}, c01 = {}, c10 = {}, c11 = {};

#define COMPUTE(buf)                                                                   \
  do {                                                                                 \
    const _Float16* Ab = &As[buf][0];                                                  \
    const _Float16* Bb = &Bs[buf][0];                                                  \
    _Pragma("unroll")                                                                  \
    for (int ks = 0; ks < 2; ++ks) {                                                   \
      int kc = ks * 32;                                                                \
      const _Float16* ap0 = Ab + (wm + nl) * LDT + kc + 8 * h;                         \
      const _Float16* ap1 = ap0 + 16 * LDT;                                            \
      v8h a0l = *(const v8h*)ap0, a0h = *(const v8h*)(ap0 + 16);                       \
      v8h a1l = *(const v8h*)ap1, a1h = *(const v8h*)(ap1 + 16);                       \
      v16h af0, af1;                                                                   \
      _Pragma("unroll")                                                                \
      for (int i = 0; i < 8; ++i) {                                                    \
        af0[i] = a0l[i]; af0[i + 8] = a0h[i];                                          \
        af1[i] = a1l[i]; af1[i + 8] = a1h[i];                                          \
      }                                                                                \
      const _Float16* bp0 = Bb + (wn + nl) * LDT + kc + 8 * h;                         \
      const _Float16* bp1 = bp0 + 16 * LDT;                                            \
      v8h b0l = *(const v8h*)bp0, b0h = *(const v8h*)(bp0 + 16);                       \
      v8h b1l = *(const v8h*)bp1, b1h = *(const v8h*)(bp1 + 16);                       \
      v16h bf0, bf1;                                                                   \
      _Pragma("unroll")                                                                \
      for (int i = 0; i < 8; ++i) {                                                    \
        bf0[i] = b0l[i]; bf0[i + 8] = b0h[i];                                          \
        bf1[i] = b1l[i]; bf1[i + 8] = b1h[i];                                          \
      }                                                                                \
      c00 = __builtin_amdgcn_wmma_f32_16x16x32_f16(false, af0, false, bf0, (short)0, c00, false, false); \
      c01 = __builtin_amdgcn_wmma_f32_16x16x32_f16(false, af0, false, bf1, (short)0, c01, false, false); \
      c10 = __builtin_amdgcn_wmma_f32_16x16x32_f16(false, af1, false, bf0, (short)0, c10, false, false); \
      c11 = __builtin_amdgcn_wmma_f32_16x16x32_f16(false, af1, false, bf1, (short)0, c11, false, false); \
    }                                                                                  \
  } while (0)

#if HAS_ASYNC_LDS
#define STAGE(buf, k0)                                            \
  do {                                                            \
    _Float16* Ad = &As[buf][0];                                   \
    _Float16* Bd = &Bs[buf][0];                                   \
    async_cp16(ag0 + (k0), Ad + ar0 * LDT + acs);                 \
    async_cp16(ag1 + (k0), Ad + ar1 * LDT + acs);                 \
    async_cp16(bg0 + (k0), Bd + (ar0     ) * LDT + acs);          \
    async_cp16(bg1 + (k0), Bd + (ar0 + 32) * LDT + acs);          \
    async_cp16(bg2 + (k0), Bd + (ar0 + 64) * LDT + acs);          \
    async_cp16(bg3 + (k0), Bd + (ar0 + 96) * LDT + acs);          \
  } while (0)

  STAGE(0, 0);
  WAIT_ASYNC();
  __syncthreads();
#pragma unroll 2
  for (int s = 0; s < NSTAGE; ++s) {
    int buf = s & 1;
    if (s + 1 < NSTAGE) STAGE(buf ^ 1, (s + 1) * KC);
    COMPUTE(buf);
    if (s + 1 < NSTAGE) {
      WAIT_ASYNC();
      __syncthreads();
    }
  }
#else
  for (int s = 0; s < NSTAGE; ++s) {
    int k0 = s * KC;
    v8h ra0 = *(const v8h*)(ag0 + k0);
    v8h ra1 = *(const v8h*)(ag1 + k0);
    v8h rb0 = *(const v8h*)(bg0 + k0);
    v8h rb1 = *(const v8h*)(bg1 + k0);
    v8h rb2 = *(const v8h*)(bg2 + k0);
    v8h rb3 = *(const v8h*)(bg3 + k0);
    __syncthreads();
    *(v8h*)&As[0][ar0 * LDT + acs]        = ra0;
    *(v8h*)&As[0][ar1 * LDT + acs]        = ra1;
    *(v8h*)&Bs[0][(ar0     ) * LDT + acs] = rb0;
    *(v8h*)&Bs[0][(ar0 + 32) * LDT + acs] = rb1;
    *(v8h*)&Bs[0][(ar0 + 64) * LDT + acs] = rb2;
    *(v8h*)&Bs[0][(ar0 + 96) * LDT + acs] = rb3;
    __syncthreads();
    COMPUTE(0);
  }
#endif

  // epilogue: out[tok, n] += w * (acc + be[e, n]); exactly 2 commutative fp adds per element
  int   cnt_loc = cnt - mb;
  int   ng0  = nb + wn + nl;
  float bev0 = be[(size_t)e * D_DIM + ng0];
  float bev1 = be[(size_t)e * D_DIM + ng0 + 16];
#pragma unroll
  for (int v = 0; v < 8; ++v) {
    int m0 = wm + 8 * h + v;
    if (m0 < cnt_loc) {
      int   tok = s_tok[m0];
      float w   = s_w[m0];
      atomicAdd(&out[(size_t)tok * D_DIM + ng0],      w * (c00[v] + bev0));
      atomicAdd(&out[(size_t)tok * D_DIM + ng0 + 16], w * (c01[v] + bev1));
    }
    int m1 = m0 + 16;
    if (m1 < cnt_loc) {
      int   tok = s_tok[m1];
      float w   = s_w[m1];
      atomicAdd(&out[(size_t)tok * D_DIM + ng0],      w * (c10[v] + bev0));
      atomicAdd(&out[(size_t)tok * D_DIM + ng0 + 16], w * (c11[v] + bev1));
    }
  }
#undef COMPUTE
#undef STAGE
}

// ---------------- launch ----------------
extern "C" void kernel_launch(void* const* d_in, const int* in_sizes, int n_in,
                              void* d_out, int out_size, void* d_ws, size_t ws_size,
                              hipStream_t stream) {
  const float* x  = (const float*)d_in[0];
  const float* Wg = (const float*)d_in[1];
  const float* bg = (const float*)d_in[2];
  const float* We = (const float*)d_in[3];
  const float* be = (const float*)d_in[4];
  float* out = (float*)d_out;

  char* ws = (char*)d_ws;
  const size_t XH_BYTES = (size_t)TOKENS * D_DIM * 2;
  const size_t WT_BYTES = (size_t)E_NUM * D_DIM * D_DIM * 2;
  _Float16* xh      = (_Float16*)ws;
  _Float16* wt      = (_Float16*)(ws + XH_BYTES);
  int*      counts  = (int*)(ws + XH_BYTES + WT_BYTES);
  int*      tok_ids = (int*)(ws + XH_BYTES + WT_BYTES + 256);
  float*    tok_w   = (float*)(ws + XH_BYTES + WT_BYTES + 256 + (size_t)E_NUM * TOKENS * 4);

  int n_out = TOKENS * D_DIM;
  moe_zero_kernel<<<1024, 256, 0, stream>>>(out, n_out, counts);

  int n4 = TOKENS * D_DIM / 4;
  moe_conv_x<<<4096, 256, 0, stream>>>((const float4*)x, (v4h*)xh, n4);

  dim3 gwe(D_DIM / 32, D_DIM / 32, E_NUM);
  dim3 bwe(32, 8);
  moe_conv_we<<<gwe, bwe, 0, stream>>>(We, wt);

  moe_gate_kernel<<<TOKENS / 4, 128, 0, stream>>>(x, Wg, bg, counts, tok_ids, tok_w);

  dim3 gg(D_DIM / BN, TOKENS / BM, E_NUM);
  moe_gemm_kernel<<<gg, 256, 0, stream>>>(xh, wt, be, counts, tok_ids, tok_w, out);
}
